// ModifiedBaseNet_9113920602557
// MI455X (gfx1250) — matrix operand order repaired and checked by reading, not compiled, not run on hardware
//
#include <hip/hip_runtime.h>

#define N_NODES 50000
#define N_EDGES 400000
#define D 512

typedef unsigned short ushort_t;
typedef __attribute__((ext_vector_type(16))) __bf16 v16bf;
typedef __attribute__((ext_vector_type(8)))  float  v8f;

union FragAB {
  v16bf v;
  uint4 q[2];
};

__device__ __forceinline__ ushort_t f2bf(float f) {
  unsigned u = __float_as_uint(f);
  u += 0x7FFFu + ((u >> 16) & 1u);   // round-to-nearest-even
  return (ushort_t)(u >> 16);
}

__device__ __forceinline__ unsigned pack2(float a, float b) {
  return (unsigned)f2bf(a) | ((unsigned)f2bf(b) << 16);
}

// ---------------------------------------------------------------- zeroing
__global__ void zero_f32(float* __restrict__ p, long n) {
  long i = (long)blockIdx.x * blockDim.x + threadIdx.x;
  if (i < n) p[i] = 0.0f;
}

// ---------------------------------------------------------------- degree
__global__ void degree_kernel(const int* __restrict__ dst, float* __restrict__ deg) {
  int e = blockIdx.x * blockDim.x + threadIdx.x;
  if (e < N_EDGES) atomicAdd(&deg[dst[e]], 1.0f);
}

// ------------------------------------------------ edge scatter (segment_sum)
__global__ void scatter_kernel(const float4* __restrict__ x,
                               const int* __restrict__ src,
                               const int* __restrict__ dst,
                               float* __restrict__ agg) {
  long gid = (long)blockIdx.x * blockDim.x + threadIdx.x;
  int e = (int)(gid >> 7);
  int c = (int)(gid & 127);
  if (e >= N_EDGES) return;
  int s = src[e], d = dst[e];
  float4 v = x[(size_t)s * (D / 4) + c];
  float* o = agg + (size_t)d * D + c * 4;
  atomicAdd(o + 0, v.x);
  atomicAdd(o + 1, v.y);
  atomicAdd(o + 2, v.z);
  atomicAdd(o + 3, v.w);
}

// -------------------------------- convert agg (with 1/deg) and x to bf16
__global__ void convert_kernel(const float4* __restrict__ agg,
                               const float4* __restrict__ xin,
                               const float* __restrict__ deg,
                               uint2* __restrict__ aggb,
                               uint2* __restrict__ xb) {
  long gid = (long)blockIdx.x * blockDim.x + threadIdx.x;
  if (gid >= (long)N_NODES * (D / 4)) return;
  int row = (int)(gid >> 7);
  float inv = 1.0f / fmaxf(deg[row], 1.0f);
  float4 a = agg[gid];
  uint2 oa;
  oa.x = pack2(a.x * inv, a.y * inv);
  oa.y = pack2(a.z * inv, a.w * inv);
  aggb[gid] = oa;
  float4 xv = xin[gid];
  uint2 ox;
  ox.x = pack2(xv.x, xv.y);
  ox.y = pack2(xv.z, xv.w);
  xb[gid] = ox;
}

// -------------------------------- transpose + convert W[k][n] -> Wt[n][k] bf16
__global__ void wt_kernel(const float* __restrict__ W, ushort_t* __restrict__ WT) {
  int i = blockIdx.x * blockDim.x + threadIdx.x;
  if (i >= D * D) return;
  int n = i / D, k = i % D;
  WT[i] = f2bf(W[(size_t)k * D + n]);
}

// -------------------------------- fused WMMA GEMM with async-LDS weight staging
// block = 256 threads (8 waves); wave -> 16(M) x 128(N) strip of C
// LDS: double-buffered {Wl,Wr} K-slices, 32(K) x 128(N) bf16, 80B-padded rows.
// K-loop unrolled x2 with NAMED fragment registers (no spillable arrays);
// B fragments software-pipelined one t-step ahead of each WMMA.
#define KSTEP 32
#define NK (D / KSTEP)        // 16 (even)
#define ROWP 40               // halves per LDS row (32 data + 8 pad)
#define MATH (128 * ROWP)     // halves per matrix slice (5120)

__global__ __launch_bounds__(256)
void gemm_kernel(const ushort_t* __restrict__ Abf,   // [N_NODES][D] bf16 (agg/deg)
                 const ushort_t* __restrict__ Xbf,   // [N_NODES][D] bf16 (x)
                 const ushort_t* __restrict__ WlT,   // [D n][D k] bf16
                 const ushort_t* __restrict__ WrT,   // [D n][D k] bf16
                 const float* __restrict__ bias,
                 float* __restrict__ out,
                 int relu) {
  __shared__ ushort_t smem[2][2][MATH];              // 40,960 bytes

  const int tid  = threadIdx.x;
  const int lane = tid & 31;
  const int wave = tid >> 5;
  const int m0 = blockIdx.x * 128 + wave * 16;       // may exceed N_NODES (tail)
  const int n0 = blockIdx.y * 128;
  const int lm = lane & 15;
  const int hi = (lane >> 4) & 1;

  // ---- async copy of one {Wl,Wr} K-slice into smem[buf] (4 ops/thread) ----
  const unsigned lds_base = (unsigned)(size_t)&smem[0][0][0];
  auto copy_slice = [&](int buf, int k) {
#pragma unroll
    for (int i = 0; i < 4; ++i) {
      int c = i * 256 + tid;
      int mat = c >> 9;
      int rem = c & 511;
      int row = rem >> 2;
      int part = rem & 3;
      const ushort_t* g = (mat ? WrT : WlT) + (size_t)(n0 + row) * D + k + part * 8;
      unsigned l = lds_base +
                   2u * (unsigned)((buf * 2 + mat) * MATH + row * ROWP + part * 8);
      asm volatile("global_load_async_to_lds_b128 %0, %1, off"
                   :: "v"(l), "v"(g) : "memory");
    }
  };

  v8f z = {};
  v8f acc[8];
#pragma unroll
  for (int t = 0; t < 8; ++t) acc[t] = z;

  const int arow = min(m0 + lm, N_NODES - 1);        // clamp tail (EXEC stays full)
  const ushort_t* aptr = Abf + (size_t)arow * D;
  const ushort_t* xptr = Xbf + (size_t)arow * D;
  const int aoff = hi ? 8 : 0;                       // A hi-lanes carry K+8 / K+24
  const int boffh = hi ? 8 : 0;                      // B hi-lanes carry K+8 / K+24

  auto load_a = [&](FragAB& a, FragAB& xf, int k) {
    a.q[0]  = *(const uint4*)(aptr + k + aoff);
    a.q[1]  = *(const uint4*)(aptr + k + aoff + 16);
    xf.q[0] = *(const uint4*)(xptr + k + aoff);
    xf.q[1] = *(const uint4*)(xptr + k + aoff + 16);
  };
  auto load_b = [&](const ushort_t* base, int t, FragAB& f) {
    const int off = (t * 16 + lm) * ROWP + boffh;
    f.q[0] = *(const uint4*)(base + off);
    f.q[1] = *(const uint4*)(base + off + 16);
  };

  FragAB fa0, fx0, fa1, fx1;

  // one K-step: consume buffer `curbuf` + (faC,fxC); prefetch into other buffer
  auto k_step = [&](int kk, int curbuf, FragAB& faC, FragAB& fxC,
                    FragAB& faN, FragAB& fxN) {
    if (kk + 1 < NK) {
      copy_slice(curbuf ^ 1, (kk + 1) * KSTEP);      // overlap next B copy
      load_a(faN, fxN, (kk + 1) * KSTEP);            // overlap next A loads
      asm volatile("s_wait_asynccnt 0x4" ::: "memory");  // buf[curbuf] done
    } else {
      asm volatile("s_wait_asynccnt 0x0" ::: "memory");
    }
    __syncthreads();                                 // all waves' writes visible

    const ushort_t* bl_s = &smem[curbuf][0][0];
    const ushort_t* br_s = &smem[curbuf][1][0];
    FragAB fl0, fr0, fl1, fr1;
    load_b(bl_s, 0, fl0);
    load_b(br_s, 0, fr0);
#pragma unroll
    for (int t = 0; t < 8; ++t) {                    // fully unrolled: refs fold
      FragAB& flC = (t & 1) ? fl1 : fl0;
      FragAB& frC = (t & 1) ? fr1 : fr0;
      FragAB& flN = (t & 1) ? fl0 : fl1;
      FragAB& frN = (t & 1) ? fr0 : fr1;
      if (t < 7) {
        load_b(bl_s, t + 1, flN);                    // B pipeline 1 step ahead
        load_b(br_s, t + 1, frN);
      }
      acc[t] = __builtin_amdgcn_wmma_f32_16x16x32_bf16(
          false, faC.v, false, flC.v, (short)0, acc[t], false, false);
      acc[t] = __builtin_amdgcn_wmma_f32_16x16x32_bf16(
          false, fxC.v, false, frC.v, (short)0, acc[t], false, false);
    }
    __syncthreads();                                 // done reading before overwrite
  };

  copy_slice(0, 0);                                  // prime B pipeline
  load_a(fa0, fx0, 0);                               // prime A pipeline

#pragma unroll 1
  for (int kk = 0; kk < NK; kk += 2) {               // x2: buffer index is constant
    k_step(kk,     0, fa0, fx0, fa1, fx1);
    k_step(kk + 1, 1, fa1, fx1, fa0, fx0);
  }

  if (m0 < N_NODES) {
#pragma unroll
    for (int t = 0; t < 8; ++t) {
      const int ncol = n0 + t * 16 + lm;
      const float bv = bias[ncol];
#pragma unroll
      for (int v = 0; v < 8; ++v) {
        const int m = m0 + v + hi * 8;
        float val = acc[t][v] + bv;
        if (relu) val = fmaxf(val, 0.0f);
        out[(size_t)m * D + ncol] = val;
      }
    }
  }
}

// ---------------------------------------------------------------- launcher
extern "C" void kernel_launch(void* const* d_in, const int* in_sizes, int n_in,
                              void* d_out, int out_size, void* d_ws, size_t ws_size,
                              hipStream_t stream) {
  (void)in_sizes; (void)n_in; (void)out_size; (void)ws_size;

  const float* x   = (const float*)d_in[0];
  const int*   ei  = (const int*)d_in[1];
  const int*   src = ei;
  const int*   dst = ei + N_EDGES;
  const float* Wl[3] = {(const float*)d_in[2], (const float*)d_in[5], (const float*)d_in[8]};
  const float* bl[3] = {(const float*)d_in[3], (const float*)d_in[6], (const float*)d_in[9]};
  const float* Wr[3] = {(const float*)d_in[4], (const float*)d_in[7], (const float*)d_in[10]};

  const size_t nd_f = (size_t)N_NODES * D * sizeof(float);     // 102,400,000
  const size_t nd_h = (size_t)N_NODES * D * sizeof(ushort_t);  //  51,200,000
  const size_t deg_b = ((size_t)N_NODES * sizeof(float) + 255) & ~(size_t)255;
  const size_t w_b   = (size_t)D * D * sizeof(ushort_t);       // 524,288

  char* ws = (char*)d_ws;
  float*    agg  = (float*)(ws);
  float*    h0   = (float*)(ws + nd_f);
  float*    h1   = (float*)(ws + 2 * nd_f);
  ushort_t* aggb = (ushort_t*)(ws + 3 * nd_f);
  ushort_t* xb   = (ushort_t*)(ws + 3 * nd_f + nd_h);
  float*    deg  = (float*)(ws + 3 * nd_f + 2 * nd_h);
  char*     wbase = ws + 3 * nd_f + 2 * nd_h + deg_b;
  ushort_t* WlT[3], * WrT[3];
  for (int l = 0; l < 3; ++l) {
    WlT[l] = (ushort_t*)(wbase + (2 * l + 0) * w_b);
    WrT[l] = (ushort_t*)(wbase + (2 * l + 1) * w_b);
  }

  const int TB = 256;

  for (int l = 0; l < 3; ++l) {
    wt_kernel<<<(D * D + TB - 1) / TB, TB, 0, stream>>>(Wl[l], WlT[l]);
    wt_kernel<<<(D * D + TB - 1) / TB, TB, 0, stream>>>(Wr[l], WrT[l]);
  }

  zero_f32<<<(N_NODES + TB - 1) / TB, TB, 0, stream>>>(deg, N_NODES);
  degree_kernel<<<(N_EDGES + TB - 1) / TB, TB, 0, stream>>>(dst, deg);

  const float* xin = x;
  float* houts[3] = {h0, h1, (float*)d_out};

  for (int l = 0; l < 3; ++l) {
    const long nagg = (long)N_NODES * D;
    zero_f32<<<(unsigned)((nagg + TB - 1) / TB), TB, 0, stream>>>(agg, nagg);

    const long nsc = (long)N_EDGES * (D / 4);
    scatter_kernel<<<(unsigned)((nsc + TB - 1) / TB), TB, 0, stream>>>(
        (const float4*)xin, src, dst, agg);

    const long ncv = (long)N_NODES * (D / 4);
    convert_kernel<<<(unsigned)((ncv + TB - 1) / TB), TB, 0, stream>>>(
        (const float4*)agg, (const float4*)xin, deg, (uint2*)aggb, (uint2*)xb);

    dim3 gg((N_NODES + 127) / 128, D / 128);
    gemm_kernel<<<gg, 256, 0, stream>>>(aggb, xb, WlT[l], WrT[l], bl[l],
                                        houts[l], (l < 2) ? 1 : 0);
    xin = houts[l];
  }
}
